// Attention_Model_70042326663883
// MI455X (gfx1250) — matrix operand order repaired and checked by reading, not compiled, not run on hardware
//
#include <hip/hip_runtime.h>
#include <hip/hip_bf16.h>

typedef float  v2f   __attribute__((ext_vector_type(2)));
typedef float  v8f   __attribute__((ext_vector_type(8)));
typedef int    v4i   __attribute__((ext_vector_type(4)));
typedef __bf16 v16bf __attribute__((ext_vector_type(16)));

#define N_NODES 8192
#define F_IN    256
#define F_OUT   128
#define LRELU_A 0.2f
#define KSTEP   64
#define P_STRIDE 68   // bf16 elems per LDS row of P (64 + 4 pad; 136B rows, dword aligned)

// ---- async-copy portability shims -----------------------------------------
typedef __attribute__((address_space(1))) v4i* gv4i_p;
typedef __attribute__((address_space(3))) v4i* lv4i_p;

#if __has_builtin(__builtin_amdgcn_global_load_async_to_lds_b128)
#define ASYNC_COPY_B128(gsrc, ldst)                                         \
    __builtin_amdgcn_global_load_async_to_lds_b128(                         \
        (gv4i_p)(gsrc), (lv4i_p)(ldst), 0, 0)
#else
#define ASYNC_COPY_B128(gsrc, ldst)                                         \
    do {                                                                    \
        unsigned _lo = (unsigned)(uintptr_t)(lv4i_p)(ldst);                 \
        unsigned long long _ga = (unsigned long long)(gsrc);                \
        asm volatile("global_load_async_to_lds_b128 %0, %1, off"            \
                     :: "v"(_lo), "v"(_ga) : "memory");                     \
    } while (0)
#endif

#if __has_builtin(__builtin_amdgcn_s_wait_asynccnt)
#define WAIT_ASYNC(n) __builtin_amdgcn_s_wait_asynccnt(n)
#else
#define WAIT_ASYNC(n) asm volatile("s_wait_asynccnt %0" :: "i"(n) : "memory")
#endif
// ---------------------------------------------------------------------------

// round-to-nearest-even f32 -> bf16 bits
__device__ __forceinline__ unsigned short f2bf(float f) {
    unsigned u = __float_as_uint(f);
    u += 0x7fffu + ((u >> 16) & 1u);
    return (unsigned short)(u >> 16);
}

// ---------------------------------------------------------------------------
// Kernel 1: h = x @ W   (f32 WMMA 16x16x4, one wave per 16x16 tile)
// ---------------------------------------------------------------------------
__global__ __launch_bounds__(32)
void k_xw(const float* __restrict__ x, const float* __restrict__ W,
          float* __restrict__ h) {
    const int r0 = blockIdx.x * 16;
    const int c0 = blockIdx.y * 16;
    const int l  = threadIdx.x;     // 0..31
    const int hl = l >> 4;          // half: 0 or 1
    const int lm = l & 15;

    v8f acc = {};
    #pragma unroll 4
    for (int k0 = 0; k0 < F_IN; k0 += 4) {
        const int ka = k0 + 2 * hl;             // this lane-group's K offset
        v2f a;                                  // A 16x4: lane<16 K={0,1}, lane>=16 K={2,3}
        a.x = x[(size_t)(r0 + lm) * F_IN + ka];
        a.y = x[(size_t)(r0 + lm) * F_IN + ka + 1];
        v2f b;                                  // B 4x16: col c0+lm, rows ka, ka+1
        b.x = W[(size_t)ka       * F_OUT + c0 + lm];
        b.y = W[(size_t)(ka + 1) * F_OUT + c0 + lm];
        acc = __builtin_amdgcn_wmma_f32_16x16x4_f32(
                  false, a, false, b, (short)0, acc, false, false);
    }
    #pragma unroll
    for (int v = 0; v < 8; ++v) {
        const int row = r0 + v + 8 * hl;        // C layout: VGPR v -> M=v (+8 upper half)
        h[(size_t)row * F_OUT + c0 + lm] = acc[v];
    }
}

// ---------------------------------------------------------------------------
// Kernel 2: s1 = h@a1, s2 = h@a2 (per row), and HT = bf16(h)^T  [F_OUT x N]
// ---------------------------------------------------------------------------
__global__ __launch_bounds__(128)
void k_prep(const float* __restrict__ h, const float* __restrict__ a,
            float* __restrict__ s1, float* __restrict__ s2,
            unsigned short* __restrict__ HT) {
    const int i = blockIdx.x;       // row
    const int c = threadIdx.x;      // 0..127 (feature)

    const float hv = h[(size_t)i * F_OUT + c];
    HT[(size_t)c * N_NODES + i] = f2bf(hv);     // transposed bf16 copy

    __shared__ float r1[128], r2[128];
    r1[c] = hv * a[c];
    r2[c] = hv * a[F_OUT + c];
    __syncthreads();
    for (int s = 64; s > 0; s >>= 1) {
        if (c < s) { r1[c] += r1[c + s]; r2[c] += r2[c + s]; }
        __syncthreads();
    }
    if (c == 0) { s1[i] = r1[0]; s2[i] = r2[0]; }
}

// ---------------------------------------------------------------------------
// Kernel 3: fused masked softmax + (att @ h) + ELU.
// 256 threads = 8 waves per block; block owns 16 output rows.
// Wave w owns output columns [16w, 16w+16).
// adj tiles (16 x 64 f32) are double-buffered into LDS with async copies.
// ---------------------------------------------------------------------------
__global__ __launch_bounds__(256)
void k_attn(const float* __restrict__ adj, const float* __restrict__ s1g,
            const float* __restrict__ s2g, const unsigned short* __restrict__ HT,
            float* __restrict__ out) {
    const int r0   = blockIdx.x * 16;
    const int tid  = threadIdx.x;
    const int wave = tid >> 5;
    const int lane = tid & 31;
    const int hl   = lane >> 4;
    const int lm   = lane & 15;

    __shared__ float sM[16], sL[16], sS1[16];
    __shared__ float sAdj[2][16 * KSTEP];            // 2 x 4 KB, b128 segments
    __shared__ unsigned short sP[16 * P_STRIDE];     // P tile, bf16

    // async tile fetch: thread t moves one 16B segment of the 16x64 tile
    const int seg_row = tid >> 4;          // 0..15
    const int seg_col = (tid & 15) << 2;   // 0,4,...,60 (floats)
    #define ISSUE_TILE(j0, buf)                                              \
        ASYNC_COPY_B128(&adj[(size_t)(r0 + seg_row) * N_NODES + (j0) + seg_col], \
                        &sAdj[(buf)][tid << 2])

    // Kick off tile 0 now; it lands while phase 1 streams adj.
    ISSUE_TILE(0, 0);

    // ---- Phase 1: per-row masked running (max, sumexp). Wave w -> rows 2w, 2w+1.
    #pragma unroll
    for (int rr = 0; rr < 2; ++rr) {
        const int i   = 2 * wave + rr;
        const int row = r0 + i;
        const float s1v = s1g[row];
        float m = -1e30f, lsum = 0.f;
        for (int j = lane; j < N_NODES; j += 32) {
            const float av = adj[(size_t)row * N_NODES + j];
            float e = s1v + s2g[j];
            e = e > 0.f ? e : LRELU_A * e;
            if (av > 0.f) {
                const float nm = fmaxf(m, e);
                lsum = lsum * __expf(m - nm) + __expf(e - nm);
                m = nm;
            }
        }
        #pragma unroll
        for (int off = 16; off > 0; off >>= 1) {
            const float om = __shfl_xor(m, off, 32);
            const float ol = __shfl_xor(lsum, off, 32);
            const float nm = fmaxf(m, om);
            lsum = lsum * __expf(m - nm) + ol * __expf(om - nm);
            m = nm;
        }
        if (lane == 0) { sM[i] = m; sL[i] = lsum; sS1[i] = s1v; }
    }
    __syncthreads();

    // ---- Phase 2: out[16 x 128] = P @ H via bf16 WMMA, K-step = 64.
    const int c0 = wave * 16;
    v8f acc = {};
    for (int j0 = 0; j0 < N_NODES; j0 += KSTEP) {
        const int buf = (j0 >> 6) & 1;
        if (j0 + KSTEP < N_NODES) {          // prefetch next tile, wait for current
            ISSUE_TILE(j0 + KSTEP, buf ^ 1);
            WAIT_ASYNC(1);
        } else {
            WAIT_ASYNC(0);
        }
        __syncthreads();                     // whole current tile visible in LDS

        // Build P tile 16x64 (bf16) in LDS: 1024 values, 4 per thread.
        {
            const int k  = tid & 63;
            const int i0 = tid >> 6;         // 0..3
            const float s2v = s2g[j0 + k];
            #pragma unroll
            for (int rep = 0; rep < 4; ++rep) {
                const int ii = i0 + (rep << 2);           // 0..15
                float e = sS1[ii] + s2v;
                e = e > 0.f ? e : LRELU_A * e;
                const float av = sAdj[buf][ii * KSTEP + k];
                const float p  = (av > 0.f) ? __expf(e - sM[ii]) : 0.f;
                sP[ii * P_STRIDE + k] = f2bf(p);
            }
        }
        __syncthreads();                     // P visible; also fences sAdj reads
                                             // before next iteration's async issue

        // Two WMMAs: K groups [0,32) and [32,64)
        #pragma unroll
        for (int kg = 0; kg < 2; ++kg) {
            // A (P tile) from LDS, contiguous k order (same as B):
            // lanes 0-15: M=lane, K {0..15}; lanes 16-31: K {16..31}
            union { v16bf v; unsigned u[8]; } A;
            const unsigned* rowp = (const unsigned*)&sP[lm * P_STRIDE];
            const int base = kg * 16 + (hl ? 8 : 0);
            #pragma unroll
            for (int v = 0; v < 4; ++v) A.u[v]     = rowp[base + v];
            #pragma unroll
            for (int v = 0; v < 4; ++v) A.u[4 + v] = rowp[base + 4 + v];
            // B (H tile, 32x16) from HT: lane = col c0+lm, 16 consecutive K per half
            union { v16bf v; unsigned u[8]; } B;
            const unsigned* colp = (const unsigned*)
                &HT[(size_t)(c0 + lm) * N_NODES + j0 + kg * 32 + 16 * hl];
            #pragma unroll
            for (int v = 0; v < 8; ++v) B.u[v] = colp[v];

            acc = __builtin_amdgcn_wmma_f32_16x16x32_bf16(
                      false, A.v, false, B.v, (short)0, acc, false, false);
        }
        // next loop iteration's top barrier separates A-reads from next P-writes
    }

    // ---- Epilogue: normalize by sumexp, ELU, store.
    #pragma unroll
    for (int v = 0; v < 8; ++v) {
        const int M = v + 8 * hl;
        float val = acc[v] / sL[M];
        val = val > 0.f ? val : (__expf(val) - 1.f);
        out[(size_t)(r0 + M) * F_OUT + c0 + lm] = val;
    }
    #undef ISSUE_TILE
}

// ---------------------------------------------------------------------------
extern "C" void kernel_launch(void* const* d_in, const int* in_sizes, int n_in,
                              void* d_out, int out_size, void* d_ws, size_t ws_size,
                              hipStream_t stream) {
    const float* x   = (const float*)d_in[0];   // [8192, 256]
    const float* adj = (const float*)d_in[1];   // [8192, 8192]
    const float* W   = (const float*)d_in[2];   // [256, 128]
    const float* a   = (const float*)d_in[3];   // [256, 1]
    float* out = (float*)d_out;                 // [8192, 128]

    char* ws = (char*)d_ws;
    float*          h  = (float*)ws;                                   // 4 MB
    unsigned short* HT = (unsigned short*)(ws + 4u * 1024 * 1024);     // 2 MB
    float*          s1 = (float*)(ws + 6u * 1024 * 1024);              // 32 KB
    float*          s2 = s1 + N_NODES;                                 // 32 KB

    dim3 g1(N_NODES / 16, F_OUT / 16);
    k_xw  <<<g1, 32, 0, stream>>>(x, W, h);
    k_prep<<<N_NODES, 128, 0, stream>>>(h, a, s1, s2, HT);
    k_attn<<<N_NODES / 16, 256, 0, stream>>>(adj, s1, s2, HT, out);
}
